// DoubleStreamBlock_86629490360823
// MI455X (gfx1250) — matrix operand order repaired and checked by reading, not compiled, not run on hardware
//
#include <hip/hip_runtime.h>
#include <math.h>

// ---------------- problem constants ----------------
#define HID   2048
#define LTXT  512
#define NIMG  2048
#define SEQ   2560          // LTXT + NIMG
#define NHEAD 16
#define HD    128
#define MLPD  8192
#define EPSV  1e-6f
#define ATT_SCALE 0.08838834764831845f   // 128^-0.5

#define USE_ASYNC_GEMM 1

typedef __bf16 bf16;
typedef __attribute__((ext_vector_type(16))) __bf16 bf16x16;
typedef __attribute__((ext_vector_type(8)))  float  f32x8;
typedef __attribute__((ext_vector_type(4)))  unsigned int u32x4;

union FragBF {          // 32 bytes: one WMMA A/B operand (16 bf16 per lane)
  bf16x16 v;
  u32x4   u[2];
};

__device__ __forceinline__ f32x8 wmma_bf16(const bf16x16& a, const bf16x16& b, f32x8 c) {
  return __builtin_amdgcn_wmma_f32_16x16x32_bf16(false, a, false, b, (short)0, c, false, false);
}

#if USE_ASYNC_GEMM
// CDNA5 async global->LDS copy (16B per lane), tracked by ASYNCcnt.
// VDST operand = per-lane LDS byte offset; flat pointers into the LDS aperture
// carry that offset in addr[31:0] (ISA 10.2 aperture mapping).
__device__ __forceinline__ void async_copy_b128(const bf16* lds_dst, const bf16* gsrc) {
  unsigned lo = (unsigned)(size_t)lds_dst;
  asm volatile("global_load_async_to_lds_b128 %0, %1, off"
               :: "v"(lo), "v"(gsrc) : "memory");
}
__device__ __forceinline__ void wait_async_le4() {
  asm volatile("s_wait_asynccnt 0x4" ::: "memory");
}
__device__ __forceinline__ void wait_async_0() {
  asm volatile("s_wait_asynccnt 0x0" ::: "memory");
}
#endif

// ---------------- elementwise helpers ----------------
__global__ void convert_f32_bf16(const float* __restrict__ x, bf16* __restrict__ y, long n) {
  long i = (long)blockIdx.x * blockDim.x + threadIdx.x;
  if (i < n) y[i] = (bf16)x[i];
}

__global__ void silu_kernel(const float* __restrict__ x, float* __restrict__ y, int n) {
  int i = blockIdx.x * blockDim.x + threadIdx.x;
  if (i < n) { float v = x[i]; y[i] = v / (1.0f + __expf(-v)); }
}

// e = silu(vec) @ W^T + b   (W: (6H,H) row-major). One wave per output.
__global__ __launch_bounds__(256) void adaln_gemv(
    const float* __restrict__ sv, const float* __restrict__ W,
    const float* __restrict__ b, float* __restrict__ e) {
  int o = blockIdx.x * 8 + (threadIdx.x >> 5);
  int l = threadIdx.x & 31;
  const float* w = W + (long)o * HID;
  float acc = 0.0f;
  for (int k = l; k < HID; k += 32) acc += sv[k] * w[k];
  #pragma unroll
  for (int off = 16; off; off >>= 1) acc += __shfl_xor(acc, off);
  if (l == 0) e[o] = acc + b[o];
}

// y_bf16 = rms_norm(x,rw) * (1+sc) + sh   -- one block per row
__global__ __launch_bounds__(256) void rms_mod_bf16(
    const float* __restrict__ x, const float* __restrict__ rw,
    const float* __restrict__ sc, const float* __restrict__ sh,
    bf16* __restrict__ y) {
  __shared__ float red[8];
  const long row = blockIdx.x;
  const int tid = threadIdx.x;
  float ss = 0.0f;
  for (int i = tid; i < HID; i += 256) { float v = x[row * HID + i]; ss += v * v; }
  #pragma unroll
  for (int off = 16; off; off >>= 1) ss += __shfl_xor(ss, off);
  if ((tid & 31) == 0) red[tid >> 5] = ss;
  __syncthreads();
  float tot = 0.0f;
  #pragma unroll
  for (int w = 0; w < 8; w++) tot += red[w];
  float norm = rsqrtf(tot / (float)HID + EPSV);
  for (int i = tid; i < HID; i += 256) {
    float v = x[row * HID + i] * norm * rw[i];
    y[row * HID + i] = (bf16)(v * (1.0f + sc[i]) + sh[i]);
  }
}

// out = x + g[col] * yv   (g per hidden column)
__global__ void gated_residual(const float* __restrict__ x, const float* __restrict__ g,
                               const float* __restrict__ yv, float* __restrict__ out, long n) {
  long i = (long)blockIdx.x * blockDim.x + threadIdx.x;
  if (i < n) { int c = (int)(i % HID); out[i] = x[i] + g[c] * yv[i]; }
}

// tanh-approx GELU, fp32 -> bf16
__global__ void gelu_bf16_kernel(const float* __restrict__ x, bf16* __restrict__ y, long n) {
  long i = (long)blockIdx.x * blockDim.x + threadIdx.x;
  if (i < n) {
    float v = x[i];
    float t = 0.7978845608028654f * (v + 0.044715f * v * v * v);
    y[i] = (bf16)(0.5f * v * (1.0f + tanhf(t)));
  }
}

// RoPE in place on img qkv (N, 6144): q at +0, k at +2048; rope (N,128) cos/sin interleaved
__global__ void rope_kernel(float* __restrict__ iqkv, const float* __restrict__ rope) {
  long i = (long)blockIdx.x * blockDim.x + threadIdx.x;   // n*NHEAD*64
  if (i >= (long)NIMG * NHEAD * 64) return;
  int p = (int)(i % 64);
  int h = (int)((i / 64) % NHEAD);
  long n = i / (64 * NHEAD);
  float c = rope[n * HD + 2 * p];
  float s = rope[n * HD + 2 * p + 1];
  #pragma unroll
  for (int qk = 0; qk < 2; qk++) {
    float* base = iqkv + n * (3 * HID) + qk * HID + h * HD + 2 * p;
    float x0 = base[0], x1 = base[1];
    base[0] = x0 * c - x1 * s;
    base[1] = x1 * c + x0 * s;
  }
}

// scatter fp32 qkv (txt 512 rows + img 2048 rows, 6144 wide) into bf16
// Qb,Kb: (head, seq, 128); Vt: (head, 128, seq)
__global__ void pack_qkv(const float* __restrict__ tq, const float* __restrict__ iq,
                         bf16* __restrict__ Qb, bf16* __restrict__ Kb, bf16* __restrict__ Vt) {
  long i = (long)blockIdx.x * blockDim.x + threadIdx.x;   // head*seq*128
  if (i >= (long)NHEAD * SEQ * HD) return;
  int d = (int)(i % HD);
  int s = (int)((i / HD) % SEQ);
  int h = (int)(i / ((long)HD * SEQ));
  const float* src = (s < LTXT) ? (tq + (long)s * (3 * HID))
                                : (iq + (long)(s - LTXT) * (3 * HID));
  float q = src[0 * HID + h * HD + d];
  float k = src[1 * HID + h * HD + d];
  float v = src[2 * HID + h * HD + d];
  Qb[((long)h * SEQ + s) * HD + d] = (bf16)q;
  Kb[((long)h * SEQ + s) * HD + d] = (bf16)k;
  Vt[((long)h * HD + d) * SEQ + s] = (bf16)v;
}

// ---------------- generic WMMA GEMM ----------------
// C[M,N](f32) = A[M,K](bf16 rowmajor) * B[N,K](bf16 rowmajor)^T (+ bias[N])
// block = 256 threads (8 waves as 2x4), tile 128x128, k-step 32.
// LDS double-buffered; tiles filled with GLOBAL_LOAD_ASYNC_TO_LDS_B128.
// Last k-step is peeled so the steady-state loop has no wait-selection branch.
#define LDSS 40   // padded bf16 row stride (keeps 16B alignment, dodges conflicts)

__global__ __launch_bounds__(256) void gemm_bf16_wmma(
    const bf16* __restrict__ A, const bf16* __restrict__ B,
    float* __restrict__ C, const float* __restrict__ bias,
    int M, int N, int K, int hasBias) {
  __shared__ __align__(16) bf16 As[2][128 * LDSS];
  __shared__ __align__(16) bf16 Bs[2][128 * LDSS];
  const int tid  = threadIdx.x;
  const int lane = tid & 31;
  const int g  = lane >> 4;
  const int ln = lane & 15;
  const int wm = (tid >> 5) >> 2;   // 0..1 : 64 rows
  const int wn = (tid >> 5) & 3;    // 0..3 : 32 cols
  const long rowBlk = (long)blockIdx.y * 128;
  const long colBlk = (long)blockIdx.x * 128;

  // per-thread chunk mapping: 512 chunks of 8 bf16 per 128x32 tile, 2 per thread
  const int r0 = tid >> 2,          c0 = (tid & 3) * 8;
  const int r1 = (tid + 256) >> 2,  c1 = ((tid + 256) & 3) * 8;

  f32x8 zero = {0.f, 0.f, 0.f, 0.f, 0.f, 0.f, 0.f, 0.f};
  f32x8 acc[4][2];
  #pragma unroll
  for (int mt = 0; mt < 4; mt++)
    #pragma unroll
    for (int nt = 0; nt < 2; nt++) acc[mt][nt] = zero;

  const int KT = K >> 5;

  // 8 WMMAs on one resident LDS buffer
  auto compute_tile = [&](int buf) {
    FragBF bfr[2];
    #pragma unroll
    for (int nt = 0; nt < 2; nt++) {            // B layout: lane=N col, K=16g+e
      const bf16* p = &Bs[buf][(wn * 32 + nt * 16 + ln) * LDSS + 16 * g];
      bfr[nt].u[0] = *(const u32x4*)p;
      bfr[nt].u[1] = *(const u32x4*)(p + 8);
    }
    #pragma unroll
    for (int mt = 0; mt < 4; mt++) {            // A layout: lane=M row, K=16g+e
      FragBF af;
      const bf16* p = &As[buf][(wm * 64 + mt * 16 + ln) * LDSS];
      af.u[0] = *(const u32x4*)(p + 16 * g);
      af.u[1] = *(const u32x4*)(p + 16 * g + 8);
      #pragma unroll
      for (int nt = 0; nt < 2; nt++)
        acc[mt][nt] = wmma_bf16(af.v, bfr[nt].v, acc[mt][nt]);
    }
  };

#if USE_ASYNC_GEMM
  // issue the 4 async tile-copies (A:2 + B:2) for k-offset k0 into buffer buf
  auto issue_tile = [&](int buf, int k0) {
    async_copy_b128(&As[buf][r0 * LDSS + c0], &A[(rowBlk + r0) * (long)K + k0 + c0]);
    async_copy_b128(&As[buf][r1 * LDSS + c1], &A[(rowBlk + r1) * (long)K + k0 + c1]);
    async_copy_b128(&Bs[buf][r0 * LDSS + c0], &B[(colBlk + r0) * (long)K + k0 + c0]);
    async_copy_b128(&Bs[buf][r1 * LDSS + c1], &B[(colBlk + r1) * (long)K + k0 + c1]);
  };
  issue_tile(0, 0);
  // steady state: always prefetch next tile, wait for current (in-order, <=4 in flight)
  for (int kt = 0; kt < KT - 1; kt++) {
    const int buf = kt & 1;
    issue_tile(buf ^ 1, (kt + 1) << 5);
    wait_async_le4();
    __syncthreads();
    compute_tile(buf);
    __syncthreads();
  }
  // peeled final tile
  wait_async_0();
  __syncthreads();
  compute_tile((KT - 1) & 1);
#else
  for (int kt = 0; kt < KT; kt++) {
    const int buf = kt & 1;
    const int k0 = kt << 5;
    *(u32x4*)&As[buf][r0 * LDSS + c0] = *(const u32x4*)&A[(rowBlk + r0) * (long)K + k0 + c0];
    *(u32x4*)&As[buf][r1 * LDSS + c1] = *(const u32x4*)&A[(rowBlk + r1) * (long)K + k0 + c1];
    *(u32x4*)&Bs[buf][r0 * LDSS + c0] = *(const u32x4*)&B[(colBlk + r0) * (long)K + k0 + c0];
    *(u32x4*)&Bs[buf][r1 * LDSS + c1] = *(const u32x4*)&B[(colBlk + r1) * (long)K + k0 + c1];
    __syncthreads();
    compute_tile(buf);
    __syncthreads();
  }
#endif

  // D layout: M = r + 8g, N = ln
  #pragma unroll
  for (int mt = 0; mt < 4; mt++) {
    #pragma unroll
    for (int nt = 0; nt < 2; nt++) {
      long col = colBlk + wn * 32 + nt * 16 + ln;
      float bv = hasBias ? bias[col] : 0.0f;
      long rbase = rowBlk + wm * 64 + mt * 16 + 8 * g;
      #pragma unroll
      for (int r = 0; r < 8; r++)
        C[(rbase + r) * (long)N + col] = acc[mt][nt][r] + bv;
    }
  }
}

// ---------------- flash attention ----------------
// one wave per (head, 32-query tile); Qb/Kb (head,seq,128), Vt (head,128,seq)
// Ob (seq, 2048) bf16 laid out [s, h*128+d]
__global__ __launch_bounds__(32) void attention_kernel(
    const bf16* __restrict__ Qb, const bf16* __restrict__ Kb,
    const bf16* __restrict__ Vt, bf16* __restrict__ Ob) {
  __shared__ __align__(16) bf16 Ps[32 * LDSS];
  const int lane = threadIdx.x;
  const int g  = lane >> 4;
  const int ln = lane & 15;
  const int h  = blockIdx.y;
  const long qbase = (long)blockIdx.x * 32;

  // resident Q fragments [mtile][kstep]
  FragBF qf[2][4];
  #pragma unroll
  for (int mt = 0; mt < 2; mt++)
    #pragma unroll
    for (int kk = 0; kk < 4; kk++) {
      const bf16* p = Qb + ((long)h * SEQ + qbase + mt * 16 + ln) * HD + kk * 32;
      qf[mt][kk].u[0] = *(const u32x4*)(p + 16 * g);
      qf[mt][kk].u[1] = *(const u32x4*)(p + 16 * g + 8);
    }

  f32x8 zero = {0.f, 0.f, 0.f, 0.f, 0.f, 0.f, 0.f, 0.f};
  f32x8 o[2][8];
  float mrow[2][8], lrow[2][8];
  #pragma unroll
  for (int mt = 0; mt < 2; mt++) {
    #pragma unroll
    for (int nt = 0; nt < 8; nt++) o[mt][nt] = zero;
    #pragma unroll
    for (int r = 0; r < 8; r++) { mrow[mt][r] = -1e30f; lrow[mt][r] = 0.0f; }
  }

  for (int kb = 0; kb < SEQ; kb += 32) {
    // S = Q K^T for this 32x32 tile
    f32x8 s[2][2];
    s[0][0] = zero; s[0][1] = zero; s[1][0] = zero; s[1][1] = zero;
    #pragma unroll
    for (int kk = 0; kk < 4; kk++) {
      FragBF kf[2];
      #pragma unroll
      for (int nt = 0; nt < 2; nt++) {   // B frag: lane = key col, K(hd) = 16g+e
        const bf16* p = Kb + ((long)h * SEQ + kb + nt * 16 + ln) * HD + kk * 32 + 16 * g;
        kf[nt].u[0] = *(const u32x4*)p;
        kf[nt].u[1] = *(const u32x4*)(p + 8);
      }
      #pragma unroll
      for (int mt = 0; mt < 2; mt++)
        #pragma unroll
        for (int nt = 0; nt < 2; nt++)
          s[mt][nt] = wmma_bf16(qf[mt][kk].v, kf[nt].v, s[mt][nt]);
    }
    // online softmax (rows live in 16-lane halves of the wave)
    #pragma unroll
    for (int mt = 0; mt < 2; mt++) {
      #pragma unroll
      for (int r = 0; r < 8; r++) {
        float s0 = s[mt][0][r] * ATT_SCALE;
        float s1 = s[mt][1][r] * ATT_SCALE;
        float v = fmaxf(s0, s1);
        #pragma unroll
        for (int off = 8; off; off >>= 1) v = fmaxf(v, __shfl_xor(v, off));
        float mnew  = fmaxf(mrow[mt][r], v);
        float alpha = __expf(mrow[mt][r] - mnew);
        float p0 = __expf(s0 - mnew);
        float p1 = __expf(s1 - mnew);
        float rs = p0 + p1;
        #pragma unroll
        for (int off = 8; off; off >>= 1) rs += __shfl_xor(rs, off);
        lrow[mt][r] = lrow[mt][r] * alpha + rs;
        mrow[mt][r] = mnew;
        #pragma unroll
        for (int nt = 0; nt < 8; nt++) o[mt][nt][r] *= alpha;
        Ps[(mt * 16 + r + 8 * g) * LDSS + ln]      = (bf16)p0;
        Ps[(mt * 16 + r + 8 * g) * LDSS + 16 + ln] = (bf16)p1;
      }
    }
    __syncthreads();
    FragBF pf[2];
    #pragma unroll
    for (int mt = 0; mt < 2; mt++) {    // A frag of P
      const bf16* p = &Ps[(mt * 16 + ln) * LDSS];
      pf[mt].u[0] = *(const u32x4*)(p + 16 * g);
      pf[mt].u[1] = *(const u32x4*)(p + 16 * g + 8);
    }
    #pragma unroll
    for (int nt = 0; nt < 8; nt++) {    // B frag of V (pre-transposed): contiguous keys
      FragBF vf;
      const bf16* p = Vt + ((long)h * HD + nt * 16 + ln) * SEQ + kb + 16 * g;
      vf.u[0] = *(const u32x4*)p;
      vf.u[1] = *(const u32x4*)(p + 8);
      #pragma unroll
      for (int mt = 0; mt < 2; mt++)
        o[mt][nt] = wmma_bf16(pf[mt].v, vf.v, o[mt][nt]);
    }
    __syncthreads();
  }
  // normalize and scatter to (seq, heads*HD)
  #pragma unroll
  for (int mt = 0; mt < 2; mt++)
    #pragma unroll
    for (int nt = 0; nt < 8; nt++)
      #pragma unroll
      for (int r = 0; r < 8; r++) {
        long q = qbase + mt * 16 + r + 8 * g;
        int  d = nt * 16 + ln;
        Ob[q * (NHEAD * HD) + h * HD + d] = (bf16)(o[mt][nt][r] / lrow[mt][r]);
      }
}

// ---------------- host-side orchestration ----------------
static inline size_t alignUp(size_t v, size_t a) { return (v + a - 1) / a * a; }
static inline long cdiv(long a, long b) { return (a + b - 1) / b; }

extern "C" void kernel_launch(void* const* d_in, const int* in_sizes, int n_in,
                              void* d_out, int out_size, void* d_ws, size_t ws_size,
                              hipStream_t stream) {
  (void)in_sizes; (void)n_in; (void)out_size; (void)ws_size;
  const float* txt        = (const float*)d_in[0];
  const float* img        = (const float*)d_in[1];
  const float* vec        = (const float*)d_in[2];
  const float* rope       = (const float*)d_in[3];
  const float* t_adaln_w  = (const float*)d_in[4];
  const float* t_adaln_b  = (const float*)d_in[5];
  const float* t_adaln_r  = (const float*)d_in[6];
  const float* i_adaln_w  = (const float*)d_in[7];
  const float* i_adaln_b  = (const float*)d_in[8];
  const float* i_adaln_r  = (const float*)d_in[9];
  const float* t_qkv_w    = (const float*)d_in[10];
  const float* i_qkv_w    = (const float*)d_in[11];
  const float* t_out_w    = (const float*)d_in[12];
  const float* i_out_w    = (const float*)d_in[13];
  const float* t_norm2    = (const float*)d_in[14];
  const float* i_norm2    = (const float*)d_in[15];
  const float* t_fc1_w    = (const float*)d_in[16];
  const float* t_fc1_b    = (const float*)d_in[17];
  const float* t_fc2_w    = (const float*)d_in[18];
  const float* t_fc2_b    = (const float*)d_in[19];
  const float* i_fc1_w    = (const float*)d_in[20];
  const float* i_fc1_b    = (const float*)d_in[21];
  const float* i_fc2_w    = (const float*)d_in[22];
  const float* i_fc2_b    = (const float*)d_in[23];
  float* out = (float*)d_out;
  float* out_txt = out;
  float* out_img = out + (long)LTXT * HID;

  // bump allocator over d_ws
  char* wsp = (char*)d_ws;
  size_t off = 0;
  auto alloc = [&](size_t bytes) -> void* {
    void* p = wsp + off;
    off = alignUp(off + bytes, 256);
    return p;
  };
  const long QKV = 3 * NHEAD * HD;   // 6144
  bf16* wqkv_t = (bf16*)alloc((size_t)QKV * HID * 2);
  bf16* wqkv_i = (bf16*)alloc((size_t)QKV * HID * 2);
  bf16* wout_t = (bf16*)alloc((size_t)HID * HID * 2);
  bf16* wout_i = (bf16*)alloc((size_t)HID * HID * 2);
  bf16* wfc1_t = (bf16*)alloc((size_t)MLPD * HID * 2);
  bf16* wfc1_i = (bf16*)alloc((size_t)MLPD * HID * 2);
  bf16* wfc2_t = (bf16*)alloc((size_t)HID * MLPD * 2);
  bf16* wfc2_i = (bf16*)alloc((size_t)HID * MLPD * 2);
  float* sv    = (float*)alloc((size_t)HID * 4);
  float* e_t   = (float*)alloc((size_t)6 * HID * 4);
  float* e_i   = (float*)alloc((size_t)6 * HID * 4);
  bf16* xn_t   = (bf16*)alloc((size_t)LTXT * HID * 2);
  bf16* xn_i   = (bf16*)alloc((size_t)NIMG * HID * 2);
  float* qkv_t = (float*)alloc((size_t)LTXT * QKV * 4);
  float* qkv_i = (float*)alloc((size_t)NIMG * QKV * 4);
  bf16* Qb     = (bf16*)alloc((size_t)NHEAD * SEQ * HD * 2);
  bf16* Kb     = (bf16*)alloc((size_t)NHEAD * SEQ * HD * 2);
  bf16* Vt     = (bf16*)alloc((size_t)NHEAD * HD * SEQ * 2);
  bf16* Ob     = (bf16*)alloc((size_t)SEQ * HID * 2);
  float* proj  = (float*)alloc((size_t)SEQ * HID * 4);
  float* fc1r  = (float*)alloc((size_t)SEQ * MLPD * 4);
  bf16* hbuf   = (bf16*)alloc((size_t)SEQ * MLPD * 2);
  float* mlpr  = (float*)alloc((size_t)SEQ * HID * 4);

  const int T = 256;
  // 1) weights -> bf16
  struct { const float* s; bf16* d; long n; } conv[8] = {
    { t_qkv_w, wqkv_t, QKV * (long)HID }, { i_qkv_w, wqkv_i, QKV * (long)HID },
    { t_out_w, wout_t, (long)HID * HID }, { i_out_w, wout_i, (long)HID * HID },
    { t_fc1_w, wfc1_t, (long)MLPD * HID }, { i_fc1_w, wfc1_i, (long)MLPD * HID },
    { t_fc2_w, wfc2_t, (long)HID * MLPD }, { i_fc2_w, wfc2_i, (long)HID * MLPD },
  };
  for (int i = 0; i < 8; i++)
    convert_f32_bf16<<<cdiv(conv[i].n, T), T, 0, stream>>>(conv[i].s, conv[i].d, conv[i].n);

  // 2) AdaLN-Zero modulation params
  silu_kernel<<<cdiv(HID, T), T, 0, stream>>>(vec, sv, HID);
  adaln_gemv<<<6 * HID / 8, T, 0, stream>>>(sv, t_adaln_w, t_adaln_b, e_t);
  adaln_gemv<<<6 * HID / 8, T, 0, stream>>>(sv, i_adaln_w, i_adaln_b, e_i);

  // 3) pre-attn modulate (sh=e+0, sc=e+H)
  rms_mod_bf16<<<LTXT, T, 0, stream>>>(txt, t_adaln_r, e_t + HID, e_t, xn_t);
  rms_mod_bf16<<<NIMG, T, 0, stream>>>(img, i_adaln_r, e_i + HID, e_i, xn_i);

  // 4) QKV projections
  gemm_bf16_wmma<<<dim3(QKV / 128, LTXT / 128), T, 0, stream>>>(
      xn_t, wqkv_t, qkv_t, nullptr, LTXT, QKV, HID, 0);
  gemm_bf16_wmma<<<dim3(QKV / 128, NIMG / 128), T, 0, stream>>>(
      xn_i, wqkv_i, qkv_i, nullptr, NIMG, QKV, HID, 0);

  // 5) RoPE on img q/k, then repack to per-head bf16 (V transposed)
  rope_kernel<<<cdiv((long)NIMG * NHEAD * 64, T), T, 0, stream>>>(qkv_i, rope);
  pack_qkv<<<cdiv((long)NHEAD * SEQ * HD, T), T, 0, stream>>>(qkv_t, qkv_i, Qb, Kb, Vt);

  // 6) joint flash attention
  attention_kernel<<<dim3(SEQ / 32, NHEAD), 32, 0, stream>>>(Qb, Kb, Vt, Ob);

  // 7) output projections
  gemm_bf16_wmma<<<dim3(HID / 128, LTXT / 128), T, 0, stream>>>(
      Ob, wout_t, proj, nullptr, LTXT, HID, HID, 0);
  gemm_bf16_wmma<<<dim3(HID / 128, NIMG / 128), T, 0, stream>>>(
      Ob + (long)LTXT * HID, wout_i, proj + (long)LTXT * HID, nullptr, NIMG, HID, HID, 0);

  // 8) gated attention residual -> d_out (g_msa = e+2H)
  gated_residual<<<cdiv((long)LTXT * HID, T), T, 0, stream>>>(
      txt, e_t + 2 * HID, proj, out_txt, (long)LTXT * HID);
  gated_residual<<<cdiv((long)NIMG * HID, T), T, 0, stream>>>(
      img, e_i + 2 * HID, proj + (long)LTXT * HID, out_img, (long)NIMG * HID);

  // 9) MLP path: norm2 modulate (sh=e+3H, sc=e+4H)
  rms_mod_bf16<<<LTXT, T, 0, stream>>>(out_txt, t_norm2, e_t + 4 * HID, e_t + 3 * HID, xn_t);
  rms_mod_bf16<<<NIMG, T, 0, stream>>>(out_img, i_norm2, e_i + 4 * HID, e_i + 3 * HID, xn_i);

  gemm_bf16_wmma<<<dim3(MLPD / 128, LTXT / 128), T, 0, stream>>>(
      xn_t, wfc1_t, fc1r, t_fc1_b, LTXT, MLPD, HID, 1);
  gemm_bf16_wmma<<<dim3(MLPD / 128, NIMG / 128), T, 0, stream>>>(
      xn_i, wfc1_i, fc1r + (long)LTXT * MLPD, i_fc1_b, NIMG, MLPD, HID, 1);

  gelu_bf16_kernel<<<cdiv((long)SEQ * MLPD, T), T, 0, stream>>>(fc1r, hbuf, (long)SEQ * MLPD);

  gemm_bf16_wmma<<<dim3(HID / 128, LTXT / 128), T, 0, stream>>>(
      hbuf, wfc2_t, mlpr, t_fc2_b, LTXT, HID, MLPD, 1);
  gemm_bf16_wmma<<<dim3(HID / 128, NIMG / 128), T, 0, stream>>>(
      hbuf + (long)LTXT * MLPD, wfc2_i, mlpr + (long)LTXT * HID, i_fc2_b, NIMG, HID, MLPD, 1);

  // 10) final gated MLP residual (g_mlp = e+5H), in place on d_out
  gated_residual<<<cdiv((long)LTXT * HID, T), T, 0, stream>>>(
      out_txt, e_t + 5 * HID, mlpr, out_txt, (long)LTXT * HID);
  gated_residual<<<cdiv((long)NIMG * HID, T), T, 0, stream>>>(
      out_img, e_i + 5 * HID, mlpr + (long)LTXT * HID, out_img, (long)NIMG * HID);
}
